// AttentionHead_197568495819
// MI455X (gfx1250) — matrix operand order repaired and checked by reading, not compiled, not run on hardware
//
#include <hip/hip_runtime.h>
#include <hip/hip_bf16.h>

// ---------------------------------------------------------------------------
// Single-head causal attention for MI455X (gfx1250, wave32, WMMA).
//   Phase 0: W (fp32) -> Wt (f16, transposed [proj][n][k])  — L2-resident
//   Phase 1: fused QKV projection, barrier-free, f16 WMMA / f32 accumulate
//   Phase 2: flash attention (online softmax), f16 WMMA, f32 accum, fp32 out
// ---------------------------------------------------------------------------

typedef __attribute__((ext_vector_type(16))) _Float16 v16h;
typedef __attribute__((ext_vector_type(8)))  _Float16 v8h;
typedef __attribute__((ext_vector_type(8)))  float    v8f;
typedef __attribute__((ext_vector_type(4)))  float    v4f;

#define NB 8
#define NL 2048
#define ND 1024
#define NH 64

// xor-shuffle within the wave via ds_swizzle (group-of-32: and=0x1f, or=0, xor=m).
// masks 1,2,4,8 stay inside each 16-lane half (rows differ between halves).
#define SWZ_XOR(v, m) \
  __int_as_float(__builtin_amdgcn_ds_swizzle(__float_as_int(v), (0x1f | ((m) << 10))))

static __device__ __forceinline__ v16h ldfrag(const _Float16* p0, const _Float16* p1) {
  v8h a = *(const v8h*)p0;
  v8h b = *(const v8h*)p1;
  return __builtin_shufflevector(a, b, 0,1,2,3,4,5,6,7,8,9,10,11,12,13,14,15);
}

static __device__ __forceinline__ v8f wmma_f16(v16h a, v16h b, v8f c) {
  // D(16x16,f32) = A(16x32,f16) * B(32x16,f16) + C
  return __builtin_amdgcn_wmma_f32_16x16x32_f16(false, a, false, b, (short)0, c,
                                                false, false);
}

// ---------------------------------------------------------------------------
// Phase 0: convert the three weight matrices to f16 in B-fragment-friendly
// transposed layout Wt[proj][n=0..63][k=0..1023]. 48 blocks x 256 threads.
// ---------------------------------------------------------------------------
__global__ void __launch_bounds__(256) wcvt_kernel(
    const float* __restrict__ wq, const float* __restrict__ wk,
    const float* __restrict__ wv, _Float16* __restrict__ Wt)
{
  const int bx   = blockIdx.x;          // 0..47
  const int proj = bx >> 4;             // 0:Q 1:K 2:V
  const int kb   = bx & 15;
  const float* W = (proj == 0) ? wq : (proj == 1) ? wk : wv;
  _Float16* dst  = Wt + (size_t)proj * NH * ND;

  const int t  = threadIdx.x;
  const int k  = kb * 64 + (t >> 2);    // k row 0..1023
  const int n0 = (t & 3) * 16;          // n base 0,16,32,48
  const float* src = W + (size_t)k * NH + n0;
  #pragma unroll
  for (int i = 0; i < 16; ++i)
    dst[(size_t)(n0 + i) * ND + k] = (_Float16)src[i];
}

// ---------------------------------------------------------------------------
// Phase 1: Q/K/V projection. grid = 3 proj * 128 mblocks, block = 256 (8 waves)
// Barrier-free: B-fragments load straight from Wt (L2/WGP$-resident) as two
// contiguous global_load_b128 per fragment. All four B-frags are issued as a
// clause BEFORE the four WMMAs so waits stagger (3,2,1,0) and overlap compute.
// Each wave: 16 rows x 64 cols. Q pre-scaled by 1/8. V stored transposed.
// ---------------------------------------------------------------------------
__global__ void __launch_bounds__(256) qkv_proj_kernel(
    const float* __restrict__ iq, const float* __restrict__ ik,
    const float* __restrict__ iv,
    const _Float16* __restrict__ Wt,
    _Float16* __restrict__ Qb, _Float16* __restrict__ Kb,
    _Float16* __restrict__ Vt)
{
  const int proj   = blockIdx.x >> 7;     // 0:Q 1:K 2:V
  const int mblock = blockIdx.x & 127;
  const int tid    = threadIdx.x;
  const int wave   = tid >> 5;
  const int lane   = tid & 31;
  const int hl     = lane >> 4;
  const int l16    = lane & 15;

  const float* A = (proj == 0) ? iq : (proj == 1) ? ik : iv;
  const _Float16* wtp = Wt + (size_t)proj * NH * ND;

  const int    mbase = mblock * 128 + wave * 16;
  const float* arow  = A + (size_t)(mbase + l16) * ND;

  const v8f zero = {0.f,0.f,0.f,0.f,0.f,0.f,0.f,0.f};
  v8f acc[4] = {zero, zero, zero, zero};

  // Per-n-tile fragment base pointers (K = hl*8 + [0,8), then +16)
  const _Float16* bbase[4];
  #pragma unroll
  for (int nt = 0; nt < 4; ++nt)
    bbase[nt] = wtp + (size_t)(nt * 16 + l16) * ND + hl * 8;

  for (int k0 = 0; k0 < ND; k0 += 32) {
    // A fragment: lane=row, halves 0-7 -> K=k0+hl*8+[0,8), 8-15 -> +16
    const float* ap = arow + k0 + hl * 8;
    __builtin_prefetch(ap + 64, 0, 1);     // next iterations' activations
    v4f x0 = *(const v4f*)ap;
    v4f x1 = *(const v4f*)(ap + 4);
    v4f x2 = *(const v4f*)(ap + 16);
    v4f x3 = *(const v4f*)(ap + 20);

    // Issue ALL B-fragment loads first (one clause, staggered waits later)
    v16h bf[4];
    #pragma unroll
    for (int nt = 0; nt < 4; ++nt) {
      const _Float16* bp = bbase[nt] + k0;
      bf[nt] = ldfrag(bp, bp + 16);
    }

    v16h afrag;
    #pragma unroll
    for (int j = 0; j < 4; ++j) {
      afrag[j]      = (_Float16)x0[j];
      afrag[4 + j]  = (_Float16)x1[j];
      afrag[8 + j]  = (_Float16)x2[j];
      afrag[12 + j] = (_Float16)x3[j];
    }

    #pragma unroll
    for (int nt = 0; nt < 4; ++nt)
      acc[nt] = wmma_f16(afrag, bf[nt], acc[nt]);
  }

  const float qscale = (proj == 0) ? 0.125f : 1.0f;   // 1/sqrt(HEAD)
  if (proj == 2) {                       // V: store transposed [b][dim][key]
    #pragma unroll
    for (int nt = 0; nt < 4; ++nt) {
      #pragma unroll
      for (int r = 0; r < 8; ++r) {
        const int rg  = mbase + r + 8 * hl;           // C layout: M = r + 8*half
        const int col = nt * 16 + l16;
        const int bb  = rg >> 11;                     // rg / NL
        const int rl  = rg & (NL - 1);
        Vt[((size_t)bb * NH + col) * NL + rl] = (_Float16)acc[nt][r];
      }
    }
  } else {
    _Float16* dst = (proj == 0) ? Qb : Kb;
    #pragma unroll
    for (int nt = 0; nt < 4; ++nt) {
      #pragma unroll
      for (int r = 0; r < 8; ++r) {
        const int rg  = mbase + r + 8 * hl;
        const int col = nt * 16 + l16;
        dst[(size_t)rg * NH + col] = (_Float16)(acc[nt][r] * qscale);
      }
    }
  }
}

// ---------------------------------------------------------------------------
// Phase 2: causal flash attention. grid = 128 blocks * 8 waves = 1024 q-tiles.
// One wave owns a 16-row q-tile; iterates 32 keys/step over the causal range
// only. K- and V-fragments for the chunk are all issued up front so the
// softmax VALU work overlaps the V loads. S kept in registers; P re-striped
// C->A layout via a per-wave LDS slice (LDS in-order per wave, no barrier).
// ---------------------------------------------------------------------------
__global__ void __launch_bounds__(256) attn_kernel(
    const _Float16* __restrict__ Qb, const _Float16* __restrict__ Kb,
    const _Float16* __restrict__ Vt, float* __restrict__ out)
{
  __shared__ _Float16 plds[8 * 16 * 32];    // per-wave 16x32 P tile (1 KB each)

  const int tid  = threadIdx.x;
  const int wave = tid >> 5;
  const int lane = tid & 31;
  const int hl   = lane >> 4;
  const int l16  = lane & 15;

  const int g     = blockIdx.x * 8 + wave;  // 0..1023
  const int b     = g >> 7;
  const int qbase = (g & 127) * 16;

  const _Float16* Qbb = Qb + (size_t)b * NL * NH;
  const _Float16* Kbb = Kb + (size_t)b * NL * NH;
  const _Float16* Vbb = Vt + (size_t)b * NH * NL;
  _Float16* pl = plds + wave * (16 * 32);

  // Q A-fragments (features 0..31 and 32..63), loaded once (already * 1/8)
  const _Float16* qrow = Qbb + (size_t)(qbase + l16) * NH;
  const v16h qa0 = ldfrag(qrow + hl * 8,      qrow + 16 + hl * 8);
  const v16h qa1 = ldfrag(qrow + 32 + hl * 8, qrow + 48 + hl * 8);

  const v8f zero = {0.f,0.f,0.f,0.f,0.f,0.f,0.f,0.f};
  v8f o[4] = {zero, zero, zero, zero};
  float mrow[8], lrow[8];
  #pragma unroll
  for (int r = 0; r < 8; ++r) { mrow[r] = -1e30f; lrow[r] = 0.f; }

  const int nk = (qbase + 47) >> 5;         // ceil((qbase+16)/32) causal chunks
  for (int t = 0; t < nk; ++t) {
    const int j0 = t * 32;

    // ---- issue all K-fragments (2 subtiles x 2 feature halves) ----
    v16h kf[4];
    #pragma unroll
    for (int s = 0; s < 2; ++s) {
      const _Float16* krow = Kbb + (size_t)(j0 + s * 16 + l16) * NH;
      kf[2 * s]     = ldfrag(krow + hl * 8,      krow + 16 + hl * 8);
      kf[2 * s + 1] = ldfrag(krow + 32 + hl * 8, krow + 48 + hl * 8);
    }
    // ---- issue all V-fragments early (consumed only after softmax) ----
    v16h vf[4];
    #pragma unroll
    for (int nt = 0; nt < 4; ++nt) {
      const _Float16* vp = Vbb + (size_t)(nt * 16 + l16) * NL + j0 + hl * 8;
      vf[nt] = ldfrag(vp, vp + 16);
    }

    // ---- S = (Q/8) K^T over 32 keys ----
    v8f S[2];
    #pragma unroll
    for (int s = 0; s < 2; ++s) {
      v8f sa = wmma_f16(qa0, kf[2 * s], zero);
      sa     = wmma_f16(qa1, kf[2 * s + 1], sa);
      const int kb = j0 + s * 16;
      if (kb + 15 > qbase) {                 // wave-uniform branch
        #pragma unroll
        for (int r = 0; r < 8; ++r) {
          const int q = qbase + r + 8 * hl;
          sa[r] = (kb + l16 > q) ? -1e30f : sa[r];
        }
      }
      S[s] = sa;
    }

    // ---- online softmax (row = 16 lanes of a half) ----
    #pragma unroll
    for (int r = 0; r < 8; ++r) {
      float rm = fmaxf(S[0][r], S[1][r]);
      rm = fmaxf(rm, SWZ_XOR(rm, 1));
      rm = fmaxf(rm, SWZ_XOR(rm, 2));
      rm = fmaxf(rm, SWZ_XOR(rm, 4));
      rm = fmaxf(rm, SWZ_XOR(rm, 8));
      const float mnew  = fmaxf(mrow[r], rm);
      const float alpha = __expf(mrow[r] - mnew);
      mrow[r] = mnew;
      const float p0 = __expf(S[0][r] - mnew);
      const float p1 = __expf(S[1][r] - mnew);
      S[0][r] = p0; S[1][r] = p1;
      float rs = p0 + p1;
      rs += SWZ_XOR(rs, 1);
      rs += SWZ_XOR(rs, 2);
      rs += SWZ_XOR(rs, 4);
      rs += SWZ_XOR(rs, 8);
      lrow[r] = lrow[r] * alpha + rs;
      #pragma unroll
      for (int nt = 0; nt < 4; ++nt) o[nt][r] *= alpha;
    }

    // ---- restripe P: C layout -> row-major LDS -> A fragment ----
    #pragma unroll
    for (int r = 0; r < 8; ++r) {
      const int rr = r + 8 * hl;
      pl[rr * 32 + l16]      = (_Float16)S[0][r];
      pl[rr * 32 + 16 + l16] = (_Float16)S[1][r];
    }
    asm volatile("" ::: "memory");   // keep ds stores before ds loads (in-order per wave)
    const _Float16* pp = pl + l16 * 32 + hl * 8;
    const v16h pa = ldfrag(pp, pp + 16);

    // ---- O += P @ V ----
    #pragma unroll
    for (int nt = 0; nt < 4; ++nt)
      o[nt] = wmma_f16(pa, vf[nt], o[nt]);
  }

  // ---- normalize and store fp32 output ----
  #pragma unroll
  for (int nt = 0; nt < 4; ++nt) {
    #pragma unroll
    for (int r = 0; r < 8; ++r) {
      const size_t orow = (size_t)b * NL + qbase + r + 8 * hl;
      out[orow * NH + nt * 16 + l16] = o[nt][r] / lrow[r];
    }
  }
}

// ---------------------------------------------------------------------------
extern "C" void kernel_launch(void* const* d_in, const int* in_sizes, int n_in,
                              void* d_out, int out_size, void* d_ws, size_t ws_size,
                              hipStream_t stream) {
  (void)in_sizes; (void)n_in; (void)out_size; (void)ws_size;
  const float* idx_k = (const float*)d_in[0];
  const float* idx_q = (const float*)d_in[1];
  const float* idx_v = (const float*)d_in[2];
  // d_in[3] = msk: causality computed analytically, not read
  const float* Wk = (const float*)d_in[4];
  const float* Wq = (const float*)d_in[5];
  const float* Wv = (const float*)d_in[6];

  _Float16* Qb = (_Float16*)d_ws;                 // [B*L, 64] f16 (pre-scaled)
  _Float16* Kb = Qb + (size_t)NB * NL * NH;       // [B*L, 64] f16
  _Float16* Vt = Kb + (size_t)NB * NL * NH;       // [B, 64, L] f16 (transposed)
  _Float16* Wt = Vt + (size_t)NB * NH * NL;       // [3, 64, 1024] f16

  wcvt_kernel<<<dim3(48), dim3(256), 0, stream>>>(Wq, Wk, Wv, Wt);

  qkv_proj_kernel<<<dim3(3 * 128), dim3(256), 0, stream>>>(
      idx_q, idx_k, idx_v, Wt, Qb, Kb, Vt);

  attn_kernel<<<dim3(128), dim3(256), 0, stream>>>(
      Qb, Kb, Vt, (float*)d_out);
}
